// Net_47090021433723
// MI455X (gfx1250) — matrix operand order repaired and checked by reading, not compiled, not run on hardware
//
#include <hip/hip_runtime.h>
#include <hip/hip_bf16.h>
#include <cstdint>

typedef _Float16 half_t;
typedef __attribute__((ext_vector_type(16))) _Float16 v16h;
typedef __attribute__((ext_vector_type(8)))  float    v8f;

#define NEGINF (-1e30f)

// ======================= WMMA helpers (wave32, 16x16x32 f16) =======================
// A fragment (16x32, f16): lane L -> row = L%16; elems 0..7 : K = k0 + (L/16)*8 + i
//                                      elems 8..15: K = k0 + 16 + (L/16)*8 + (i-8)
// B fragment (32x16, f16): lane L -> col = L%16; same K order as A fragment
//   (weights pre-packed transposed [N, Kpad] so this is contiguous per lane)
// C/D (16x16 f32): lane L -> col = L%16, vgpr r -> row = r + 8*(L/16)

__device__ __forceinline__ v8f wave_tile_f16(const half_t* __restrict__ aTile, int aStride,
                                             const half_t* __restrict__ wtRow, int kLen,
                                             int lane, v8f acc) {
  const int lh = lane & 15;
  const int hf = lane >> 4;
  const half_t* ar = aTile + (size_t)lh * aStride;
  for (int k0 = 0; k0 < kLen; k0 += 32) {
    v16h a, b;
#pragma unroll
    for (int i = 0; i < 8; ++i) {
      a[i]     = ar[k0 + hf * 8 + i];
      a[i + 8] = ar[k0 + 16 + hf * 8 + i];
    }
#pragma unroll
    for (int i = 0; i < 8; ++i) {
      b[i]     = wtRow[k0 + hf * 8 + i];
      b[i + 8] = wtRow[k0 + 16 + hf * 8 + i];
    }
    acc = __builtin_amdgcn_wmma_f32_16x16x32_f16(false, a, false, b, (short)0, acc,
                                                 false, false);
  }
  return acc;
}

template <typename T>
__device__ __forceinline__ void store_tile(T* __restrict__ outRow0, int stride, v8f acc,
                                           const float* __restrict__ bias, int nBase,
                                           int nReal, bool relu, int lane) {
  const int lh = lane & 15, hf = lane >> 4;
  const int n = nBase + lh;
  float bv = (bias && n < nReal) ? bias[n] : 0.0f;
#pragma unroll
  for (int r = 0; r < 8; ++r) {
    float v = acc[r] + bv;
    if (relu && v < 0.0f) v = 0.0f;
    outRow0[(size_t)(r + hf * 8) * stride + n] = (T)v;
  }
}

// ======================= Generic GEMM: out = act(A @ WT^T + b) =====================
// A: [M, KP] f16 (KP mult of 32, M mult of 64), WT: [Npad, KP] f16, out: [M, Npad].
// block = 256 threads = 8 waves; block covers 64 rows (4 M-tiles); wave w owns one
// N-tile and reuses its B fragment across all 4 M-tiles (4 WMMAs per B load).
// A staged to LDS in 256-col K chunks via GLOBAL_LOAD_ASYNC_TO_LDS_B128 (ASYNCcnt).
template <typename OutT>
__global__ void gemm_kernel(const half_t* __restrict__ A, const half_t* __restrict__ WT,
                            const float* __restrict__ bias, OutT* __restrict__ out,
                            int M, int KP, int Npad, int Nreal, int relu) {
  __shared__ __align__(16) half_t sA[64 * 256];
  const int tid = threadIdx.x, lane = tid & 31, wave = tid >> 5;
  const int lh = lane & 15, hf = lane >> 4;
  const int m0 = blockIdx.x * 64;
  const int NT = Npad >> 4;
  const int nt = blockIdx.y * 8 + wave;
  const bool active = nt < NT;
  const half_t* wtn = WT + (size_t)((active ? nt : 0) * 16 + lh) * KP;
  __builtin_prefetch(wtn, 0, 0);
  v8f acc0 = {}, acc1 = {}, acc2 = {}, acc3 = {};

  for (int kc = 0; kc < KP; kc += 256) {
    const int kw = min(256, KP - kc);
    // async-stage A[m0:m0+64, kc:kc+kw] -> sA[64, kw] (16B per lane per op)
    for (int i = tid * 8; i < 64 * kw; i += 256 * 8) {
      const int r = i / kw, ccol = i - r * kw;
      const unsigned lm = (unsigned)(size_t)(sA + i);
      const unsigned long long ga =
          (unsigned long long)(size_t)(A + (size_t)(m0 + r) * KP + kc + ccol);
      asm volatile("global_load_async_to_lds_b128 %0, %1, off"
                   :: "v"(lm), "v"(ga) : "memory");
    }
    asm volatile("s_wait_asynccnt 0x0" ::: "memory");
    __syncthreads();
    if (active) {
      for (int k0 = 0; k0 < kw; k0 += 32) {
        v16h b;
#pragma unroll
        for (int i = 0; i < 8; ++i) {
          b[i]     = wtn[kc + k0 + hf * 8 + i];
          b[i + 8] = wtn[kc + k0 + 16 + hf * 8 + i];
        }
#pragma unroll
        for (int mt = 0; mt < 4; ++mt) {
          const half_t* ar = sA + (size_t)(mt * 16 + lh) * kw;
          v16h a;
#pragma unroll
          for (int i = 0; i < 8; ++i) {
            a[i]     = ar[k0 + hf * 8 + i];
            a[i + 8] = ar[k0 + 16 + hf * 8 + i];
          }
          v8f& acc = (mt == 0) ? acc0 : (mt == 1) ? acc1 : (mt == 2) ? acc2 : acc3;
          acc = __builtin_amdgcn_wmma_f32_16x16x32_f16(false, a, false, b, (short)0, acc,
                                                       false, false);
        }
      }
    }
    __syncthreads();
  }
  if (active) {
    store_tile<OutT>(out + (size_t)(m0 +  0) * Npad, Npad, acc0, bias, nt * 16, Nreal, relu != 0, lane);
    store_tile<OutT>(out + (size_t)(m0 + 16) * Npad, Npad, acc1, bias, nt * 16, Nreal, relu != 0, lane);
    store_tile<OutT>(out + (size_t)(m0 + 32) * Npad, Npad, acc2, bias, nt * 16, Nreal, relu != 0, lane);
    store_tile<OutT>(out + (size_t)(m0 + 48) * Npad, Npad, acc3, bias, nt * 16, Nreal, relu != 0, lane);
  }
}

// ================== Fused PointNetConv (gather -> 3-layer MLP -> masked max) =======
// One block (256 thr / 8 waves) per center. All activations live in LDS.
// LDS regions aliased by liveness:  region1 = {sIn | sH2}, region2 = {sH1 | sOut}
template <int KIN, int H1, int H2, int H3, int F>
__global__ void sa_conv_kernel(const half_t* __restrict__ xsrc, int xStride,
                               const float* __restrict__ pos_src,
                               const float* __restrict__ pos_c,
                               const int* __restrict__ nei, const int* __restrict__ cnt,
                               const half_t* __restrict__ wt1, const float* __restrict__ b1, int n1r,
                               const half_t* __restrict__ wt2, const float* __restrict__ b2, int n2r,
                               const half_t* __restrict__ wt3, const float* __restrict__ b3, int n3r,
                               half_t* __restrict__ out, int centersPerGraph, int srcPerGraph) {
  constexpr int R1B = 64 * ((KIN > H2) ? KIN : H2) * 2;
  constexpr int R2B = 64 * ((H1 > H3) ? H1 : H3) * 2;
  __shared__ __align__(16) char pool[R1B + R2B];
  half_t* sIn  = (half_t*)pool;
  half_t* sH2  = (half_t*)pool;
  half_t* sH1  = (half_t*)(pool + R1B);
  half_t* sOut = (half_t*)(pool + R1B);

  const int c = blockIdx.x;
  const int g = c / centersPerGraph;
  const int tid = threadIdx.x, lane = tid & 31, wave = tid >> 5;
  const int nvalid = min(cnt[c], 64);
  const float cx = pos_c[c * 3 + 0], cy = pos_c[c * 3 + 1], cz = pos_c[c * 3 + 2];

  // gather+concat [x_j, p_j - p_i] into sIn [64, KIN], zero padded
  for (int i = tid; i < 64 * KIN; i += 256) {
    const int n = i / KIN, k = i - n * KIN;
    half_t v = (half_t)0.0f;
    if (n < nvalid) {
      const int j = g * srcPerGraph + nei[(size_t)c * 64 + n];
      if (k < F) {
        v = xsrc[(size_t)j * xStride + k];
      } else if (k < F + 3) {
        const int a = k - F;
        const float pc = (a == 0) ? cx : ((a == 1) ? cy : cz);
        v = (half_t)(pos_src[(size_t)j * 3 + a] - pc);
      }
    }
    sIn[i] = v;
  }
  __syncthreads();

  {  // layer 1: sIn[64,KIN] -> sH1[64,H1], relu
    const int NT = H1 >> 4;
    for (int t = wave; t < 4 * NT; t += 8) {
      const int mt = t / NT, nt = t - mt * NT;
      const half_t* wtn = wt1 + (size_t)(nt * 16 + (lane & 15)) * KIN;
      v8f acc = {};
      acc = wave_tile_f16(sIn + mt * 16 * KIN, KIN, wtn, KIN, lane, acc);
      store_tile<half_t>(sH1 + mt * 16 * H1, H1, acc, b1, nt * 16, n1r, true, lane);
    }
  }
  __syncthreads();
  {  // layer 2: sH1[64,H1] -> sH2[64,H2], relu
    const int NT = H2 >> 4;
    for (int t = wave; t < 4 * NT; t += 8) {
      const int mt = t / NT, nt = t - mt * NT;
      const half_t* wtn = wt2 + (size_t)(nt * 16 + (lane & 15)) * H1;
      v8f acc = {};
      acc = wave_tile_f16(sH1 + mt * 16 * H1, H1, wtn, H1, lane, acc);
      store_tile<half_t>(sH2 + mt * 16 * H2, H2, acc, b2, nt * 16, n2r, true, lane);
    }
  }
  __syncthreads();
  {  // layer 3: sH2[64,H2] -> sOut[64,H3], plain (max-aggregated next)
    const int NT = H3 >> 4;
    for (int t = wave; t < 4 * NT; t += 8) {
      const int mt = t / NT, nt = t - mt * NT;
      const half_t* wtn = wt3 + (size_t)(nt * 16 + (lane & 15)) * H2;
      v8f acc = {};
      acc = wave_tile_f16(sH2 + mt * 16 * H2, H2, wtn, H2, lane, acc);
      store_tile<half_t>(sOut + mt * 16 * H3, H3, acc, b3, nt * 16, n3r, false, lane);
    }
  }
  __syncthreads();
  // masked max over valid neighbor rows
  for (int j = tid; j < H3; j += 256) {
    float m = NEGINF;
    for (int n = 0; n < nvalid; ++n) m = fmaxf(m, (float)sOut[n * H3 + j]);
    out[(size_t)c * H3 + j] = (half_t)m;
  }
}

// ======================= Farthest point sampling (one block / graph) ===============
__global__ void fps_kernel(const float* __restrict__ pos, int Npts, int nsample,
                           int* __restrict__ idx) {
  __shared__ float mind[4000];
  __shared__ float rv[256];
  __shared__ int ri[256];
  __shared__ int sLast;
  const int g = blockIdx.x, tid = threadIdx.x;
  const float* p = pos + (size_t)g * Npts * 3;
  for (int i = tid; i < Npts; i += 256) mind[i] = 1e30f;
  if (tid == 0) { idx[(size_t)g * nsample] = 0; sLast = 0; }
  __syncthreads();
  int last = 0;
  for (int s = 1; s < nsample; ++s) {
    const float lx = p[last * 3], ly = p[last * 3 + 1], lz = p[last * 3 + 2];
    float bv = -1.0f;
    int bi = 0x7fffffff;
    for (int i = tid; i < Npts; i += 256) {
      const float dx = p[i * 3] - lx, dy = p[i * 3 + 1] - ly, dz = p[i * 3 + 2] - lz;
      const float d = dx * dx + dy * dy + dz * dz;
      const float m = fminf(mind[i], d);
      mind[i] = m;
      if (m > bv) { bv = m; bi = i; }
    }
    rv[tid] = bv; ri[tid] = bi;
    __syncthreads();
    for (int off = 128; off; off >>= 1) {
      if (tid < off) {
        const float ov = rv[tid + off]; const int oi = ri[tid + off];
        if (ov > rv[tid] || (ov == rv[tid] && oi < ri[tid])) { rv[tid] = ov; ri[tid] = oi; }
      }
      __syncthreads();
    }
    if (tid == 0) { idx[(size_t)g * nsample + s] = ri[0]; sLast = ri[0]; }
    __syncthreads();
    last = sLast;
    __syncthreads();
  }
}

__global__ void gather_pos_kernel(const float* __restrict__ src, const int* __restrict__ idx,
                                  float* __restrict__ dst, int n, int srcN, int total) {
  const int t = blockIdx.x * blockDim.x + threadIdx.x;
  if (t >= total) return;
  const int c = t % 3, r = t / 3, g = r / n, j = r - g * n;
  dst[t] = src[((size_t)g * srcN + idx[(size_t)g * n + j]) * 3 + c];
}

// ======================= Radius neighbors (one wave / center, index order) =========
__global__ void radius_kernel(const float* __restrict__ pos_src, const float* __restrict__ pos_c,
                              int* __restrict__ nei, int* __restrict__ cnt, int Mc,
                              int centersPerGraph, int srcPerGraph, float r2) {
  const int wave = threadIdx.x >> 5, lane = threadIdx.x & 31;
  const int c = blockIdx.x * 8 + wave;
  if (c >= Mc) return;
  const int g = c / centersPerGraph;
  const float* ps = pos_src + (size_t)g * srcPerGraph * 3;
  const float cx = pos_c[c * 3], cy = pos_c[c * 3 + 1], cz = pos_c[c * 3 + 2];
  int count = 0;
  for (int base = 0; base < srcPerGraph && count < 64; base += 32) {
    const int i = base + lane;
    bool val = false;
    if (i < srcPerGraph) {
      const float dx = ps[i * 3] - cx, dy = ps[i * 3 + 1] - cy, dz = ps[i * 3 + 2] - cz;
      val = (dx * dx + dy * dy + dz * dz) <= r2;
    }
    const unsigned long long m = __ballot(val);
    const int pre = (int)__popcll(m & ((1ull << lane) - 1ull));
    if (val && (count + pre) < 64) nei[(size_t)c * 64 + count + pre] = i;
    count += (int)__popcll(m);
  }
  count = min(count, 64);
  if (lane == 0) cnt[c] = count;
  for (int j = count + lane; j < 64; j += 32) nei[(size_t)c * 64 + j] = 0;
}

// ======================= kNN (k=3) + inverse-d2 weights ============================
__global__ void knn3_kernel(const float* __restrict__ pos_src, const float* __restrict__ pos_dst,
                            int* __restrict__ idx, float* __restrict__ w, int Nd,
                            int dstPerGraph, int srcPerGraph) {
  const int d = blockIdx.x * blockDim.x + threadIdx.x;
  if (d >= Nd) return;
  const int g = d / dstPerGraph;
  const float* ps = pos_src + (size_t)g * srcPerGraph * 3;
  const float dx0 = pos_dst[d * 3], dy0 = pos_dst[d * 3 + 1], dz0 = pos_dst[d * 3 + 2];
  float b0 = 1e30f, b1 = 1e30f, b2 = 1e30f;
  int i0 = 0, i1 = 0, i2 = 0;
  for (int i = 0; i < srcPerGraph; ++i) {
    const float dx = ps[i * 3] - dx0, dy = ps[i * 3 + 1] - dy0, dz = ps[i * 3 + 2] - dz0;
    const float d2 = dx * dx + dy * dy + dz * dz;
    if (d2 < b0) { b2 = b1; i2 = i1; b1 = b0; i1 = i0; b0 = d2; i0 = i; }
    else if (d2 < b1) { b2 = b1; i2 = i1; b1 = d2; i1 = i; }
    else if (d2 < b2) { b2 = d2; i2 = i; }
  }
  float w0 = 1.0f / fmaxf(b0, 1e-16f);
  float w1 = 1.0f / fmaxf(b1, 1e-16f);
  float w2 = 1.0f / fmaxf(b2, 1e-16f);
  const float s = 1.0f / (w0 + w1 + w2);
  idx[d * 3 + 0] = g * srcPerGraph + i0;
  idx[d * 3 + 1] = g * srcPerGraph + i1;
  idx[d * 3 + 2] = g * srcPerGraph + i2;
  w[d * 3 + 0] = w0 * s; w[d * 3 + 1] = w1 * s; w[d * 3 + 2] = w2 * s;
}

// interp(xs via 3-NN) ++ skip, zero-padded to KP
__global__ void interp_concat_kernel(const half_t* __restrict__ xs, int Fs,
                                     const int* __restrict__ idx, const float* __restrict__ w,
                                     const half_t* __restrict__ skip, int Fk,
                                     half_t* __restrict__ out, long long total, int KP) {
  const long long t = (long long)blockIdx.x * blockDim.x + threadIdx.x;
  if (t >= total) return;
  const long long d = t / KP;
  const int k = (int)(t - d * KP);
  float v = 0.0f;
  if (k < Fs) {
    const long long b = d * 3;
    v = w[b + 0] * (float)xs[(size_t)idx[b + 0] * Fs + k] +
        w[b + 1] * (float)xs[(size_t)idx[b + 1] * Fs + k] +
        w[b + 2] * (float)xs[(size_t)idx[b + 2] * Fs + k];
  } else if (k < Fs + Fk) {
    v = (float)skip[d * Fk + (k - Fs)];
  }
  out[t] = (half_t)v;
}

// sa3 input: [x2(256) ++ pos2(3)] padded to 288
__global__ void concat_pos_kernel(const half_t* __restrict__ x2, const float* __restrict__ pos2,
                                  half_t* __restrict__ out, int total) {
  const int t = blockIdx.x * blockDim.x + threadIdx.x;
  if (t >= total) return;
  const int d = t / 288, k = t - d * 288;
  float v = 0.0f;
  if (k < 256) v = (float)x2[(size_t)d * 256 + k];
  else if (k < 259) v = pos2[(size_t)d * 3 + (k - 256)];
  out[t] = (half_t)v;
}

// fp3 input: [xG(1024, per graph, k=1 interp) ++ x2(256)] = 1280
__global__ void fp3_concat_kernel(const half_t* __restrict__ xG, const half_t* __restrict__ x2,
                                  half_t* __restrict__ out, int total, int n2) {
  const int t = blockIdx.x * blockDim.x + threadIdx.x;
  if (t >= total) return;
  const int d = t / 1280, k = t - d * 1280;
  const int g = d / n2;
  out[t] = (k < 1024) ? xG[(size_t)g * 1024 + k] : x2[(size_t)d * 256 + (k - 1024)];
}

__global__ void maxpool_kernel(const half_t* __restrict__ h, half_t* __restrict__ xG,
                               int rows, int Fdim, int total) {
  const int t = blockIdx.x * blockDim.x + threadIdx.x;
  if (t >= total) return;
  const int g = t / Fdim, f = t - g * Fdim;
  float m = NEGINF;
  for (int r = 0; r < rows; ++r) m = fmaxf(m, (float)h[((size_t)g * rows + r) * Fdim + f]);
  xG[t] = (half_t)m;
}

// weight pack: W f32 [Kin, Nout] -> WT f16 [Npad, KP] (transposed, zero padded)
__global__ void pack_wt_kernel(const float* __restrict__ W, int Kin, int Nout,
                               half_t* __restrict__ WT, int KP, int Npad, int total) {
  const int t = blockIdx.x * blockDim.x + threadIdx.x;
  if (t >= total) return;
  const int n = t / KP, k = t - n * KP;
  WT[t] = (n < Nout && k < Kin) ? (half_t)W[(size_t)k * Nout + n] : (half_t)0.0f;
}

__global__ void cvt_half_kernel(const float* __restrict__ in, half_t* __restrict__ out, int total) {
  const int t = blockIdx.x * blockDim.x + threadIdx.x;
  if (t < total) out[t] = (half_t)in[t];
}

__global__ void logsoftmax_kernel(const float* __restrict__ logits, float* __restrict__ out,
                                  int M) {
  const int r = blockIdx.x * blockDim.x + threadIdx.x;
  if (r >= M) return;
  const float* l = logits + (size_t)r * 16;
  float mx = l[0];
  for (int c = 1; c < 10; ++c) mx = fmaxf(mx, l[c]);
  float s = 0.0f;
  for (int c = 0; c < 10; ++c) s += __expf(l[c] - mx);
  const float ls = __logf(s);
  for (int c = 0; c < 10; ++c) out[(size_t)r * 10 + c] = l[c] - mx - ls;
}

// ===================================================================================
extern "C" void kernel_launch(void* const* d_in, const int* in_sizes, int n_in,
                              void* d_out, int out_size, void* d_ws, size_t ws_size,
                              hipStream_t stream) {
  constexpr int B = 16, NPTS = 4000, N1 = 800, N2 = 200;
  const float* x   = (const float*)d_in[0];
  const float* pos = (const float*)d_in[1];

  char* wp = (char*)d_ws;
  auto alloc = [&](size_t bytes) -> void* {
    void* p = (void*)wp;
    wp += (bytes + 255) & ~(size_t)255;
    return p;
  };

  // ---- packed weights (transposed f16, K padded to 32, N padded to 16) ----
  struct LW { half_t* wt; const float* b; int KP, Npad, Nreal; };
  auto packW = [&](int widx, int Kin, int Nout, int KP, int Npad) -> LW {
    half_t* wt = (half_t*)alloc((size_t)Npad * KP * sizeof(half_t));
    const int total = Npad * KP;
    pack_wt_kernel<<<(total + 255) / 256, 256, 0, stream>>>(
        (const float*)d_in[widx], Kin, Nout, wt, KP, Npad, total);
    return LW{wt, (const float*)d_in[widx + 1], KP, Npad, Nout};
  };

  LW sa1l1 = packW(3, 6, 64, 32, 64),    sa1l2 = packW(5, 64, 64, 64, 64),
     sa1l3 = packW(7, 64, 128, 64, 128);
  LW sa2l1 = packW(9, 131, 128, 160, 128), sa2l2 = packW(11, 128, 128, 128, 128),
     sa2l3 = packW(13, 128, 256, 128, 256);
  LW sa3l1 = packW(15, 259, 256, 288, 256), sa3l2 = packW(17, 256, 512, 256, 512),
     sa3l3 = packW(19, 512, 1024, 512, 1024);
  LW fp3l1 = packW(21, 1280, 256, 1280, 256), fp3l2 = packW(23, 256, 256, 256, 256);
  LW fp2l1 = packW(25, 384, 256, 384, 256),   fp2l2 = packW(27, 256, 128, 256, 128);
  LW fp1l1 = packW(29, 131, 128, 160, 128),   fp1l2 = packW(31, 128, 128, 128, 128),
     fp1l3 = packW(33, 128, 128, 128, 128);
  LW hd1 = packW(35, 128, 128, 128, 128), hd2 = packW(37, 128, 128, 128, 128),
     hd3 = packW(39, 128, 10, 128, 16);

  // ---- activations / scratch ----
  half_t* x0h   = (half_t*)alloc((size_t)B * NPTS * 3 * sizeof(half_t));
  int*    idx1  = (int*)alloc((size_t)B * N1 * sizeof(int));
  float*  pos1  = (float*)alloc((size_t)B * N1 * 3 * sizeof(float));
  int*    nei1  = (int*)alloc((size_t)B * N1 * 64 * sizeof(int));
  int*    cnt1  = (int*)alloc((size_t)B * N1 * sizeof(int));
  half_t* x1    = (half_t*)alloc((size_t)B * N1 * 128 * sizeof(half_t));
  int*    idx2  = (int*)alloc((size_t)B * N2 * sizeof(int));
  float*  pos2  = (float*)alloc((size_t)B * N2 * 3 * sizeof(float));
  int*    nei2  = (int*)alloc((size_t)B * N2 * 64 * sizeof(int));
  int*    cnt2  = (int*)alloc((size_t)B * N2 * sizeof(int));
  half_t* x2    = (half_t*)alloc((size_t)B * N2 * 256 * sizeof(half_t));
  half_t* sa3in = (half_t*)alloc((size_t)B * N2 * 288 * sizeof(half_t));
  half_t* sa3h1 = (half_t*)alloc((size_t)B * N2 * 256 * sizeof(half_t));
  half_t* sa3h2 = (half_t*)alloc((size_t)B * N2 * 512 * sizeof(half_t));
  half_t* sa3h3 = (half_t*)alloc((size_t)B * N2 * 1024 * sizeof(half_t));
  half_t* xG    = (half_t*)alloc((size_t)B * 1024 * sizeof(half_t));
  half_t* fp3in = (half_t*)alloc((size_t)B * N2 * 1280 * sizeof(half_t));
  half_t* fp3h  = (half_t*)alloc((size_t)B * N2 * 256 * sizeof(half_t));
  half_t* fp3o  = (half_t*)alloc((size_t)B * N2 * 256 * sizeof(half_t));
  int*    k2i   = (int*)alloc((size_t)B * N1 * 3 * sizeof(int));
  float*  k2w   = (float*)alloc((size_t)B * N1 * 3 * sizeof(float));
  half_t* fp2in = (half_t*)alloc((size_t)B * N1 * 384 * sizeof(half_t));
  half_t* fp2h  = (half_t*)alloc((size_t)B * N1 * 256 * sizeof(half_t));
  half_t* fp2o  = (half_t*)alloc((size_t)B * N1 * 128 * sizeof(half_t));
  int*    k1i   = (int*)alloc((size_t)B * NPTS * 3 * sizeof(int));
  float*  k1w   = (float*)alloc((size_t)B * NPTS * 3 * sizeof(float));
  half_t* fp1in = (half_t*)alloc((size_t)B * NPTS * 160 * sizeof(half_t));
  half_t* fp1a  = (half_t*)alloc((size_t)B * NPTS * 128 * sizeof(half_t));
  half_t* fp1b  = (half_t*)alloc((size_t)B * NPTS * 128 * sizeof(half_t));
  half_t* fp1c  = (half_t*)alloc((size_t)B * NPTS * 128 * sizeof(half_t));
  half_t* hh1   = (half_t*)alloc((size_t)B * NPTS * 128 * sizeof(half_t));
  half_t* hh2   = (half_t*)alloc((size_t)B * NPTS * 128 * sizeof(half_t));
  float*  logit = (float*)alloc((size_t)B * NPTS * 16 * sizeof(float));

  auto gemmH = [&](const half_t* A, const LW& L, half_t* out, int M, int relu) {
    dim3 g(M / 64, (L.Npad / 16 + 7) / 8);
    gemm_kernel<half_t><<<g, 256, 0, stream>>>(A, L.wt, L.b, out, M, L.KP, L.Npad,
                                               L.Nreal, relu);
  };
  auto gemmF = [&](const half_t* A, const LW& L, float* out, int M, int relu) {
    dim3 g(M / 64, (L.Npad / 16 + 7) / 8);
    gemm_kernel<float><<<g, 256, 0, stream>>>(A, L.wt, L.b, out, M, L.KP, L.Npad,
                                              L.Nreal, relu);
  };

  // x -> f16
  {
    const int tot = B * NPTS * 3;
    cvt_half_kernel<<<(tot + 255) / 256, 256, 0, stream>>>(x, x0h, tot);
  }

  // ---- SA1 ----
  fps_kernel<<<B, 256, 0, stream>>>(pos, NPTS, N1, idx1);
  {
    const int tot = B * N1 * 3;
    gather_pos_kernel<<<(tot + 255) / 256, 256, 0, stream>>>(pos, idx1, pos1, N1, NPTS, tot);
  }
  radius_kernel<<<(B * N1 + 7) / 8, 256, 0, stream>>>(pos, pos1, nei1, cnt1, B * N1, N1,
                                                      NPTS, 0.2f * 0.2f);
  sa_conv_kernel<32, 64, 64, 128, 3><<<B * N1, 256, 0, stream>>>(
      x0h, 3, pos, pos1, nei1, cnt1, sa1l1.wt, sa1l1.b, 64, sa1l2.wt, sa1l2.b, 64,
      sa1l3.wt, sa1l3.b, 128, x1, N1, NPTS);

  // ---- SA2 ----
  fps_kernel<<<B, 256, 0, stream>>>(pos1, N1, N2, idx2);
  {
    const int tot = B * N2 * 3;
    gather_pos_kernel<<<(tot + 255) / 256, 256, 0, stream>>>(pos1, idx2, pos2, N2, N1, tot);
  }
  radius_kernel<<<(B * N2 + 7) / 8, 256, 0, stream>>>(pos1, pos2, nei2, cnt2, B * N2, N2,
                                                      N1, 0.4f * 0.4f);
  sa_conv_kernel<160, 128, 128, 256, 128><<<B * N2, 256, 0, stream>>>(
      x1, 128, pos1, pos2, nei2, cnt2, sa2l1.wt, sa2l1.b, 128, sa2l2.wt, sa2l2.b, 128,
      sa2l3.wt, sa2l3.b, 256, x2, N2, N1);

  // ---- global SA ----
  {
    const int tot = B * N2 * 288;
    concat_pos_kernel<<<(tot + 255) / 256, 256, 0, stream>>>(x2, pos2, sa3in, tot);
  }
  gemmH(sa3in, sa3l1, sa3h1, B * N2, 1);
  gemmH(sa3h1, sa3l2, sa3h2, B * N2, 1);
  gemmH(sa3h2, sa3l3, sa3h3, B * N2, 0);
  {
    const int tot = B * 1024;
    maxpool_kernel<<<(tot + 255) / 256, 256, 0, stream>>>(sa3h3, xG, N2, 1024, tot);
  }

  // ---- FP3 (k=1 global interp == broadcast) ----
  {
    const int tot = B * N2 * 1280;
    fp3_concat_kernel<<<(tot + 255) / 256, 256, 0, stream>>>(xG, x2, fp3in, tot, N2);
  }
  gemmH(fp3in, fp3l1, fp3h, B * N2, 1);
  gemmH(fp3h, fp3l2, fp3o, B * N2, 0);

  // ---- FP2 (k=3, pos2 -> pos1) ----
  knn3_kernel<<<(B * N1 + 255) / 256, 256, 0, stream>>>(pos2, pos1, k2i, k2w, B * N1, N1, N2);
  {
    const long long tot = (long long)B * N1 * 384;
    interp_concat_kernel<<<(int)((tot + 255) / 256), 256, 0, stream>>>(
        fp3o, 256, k2i, k2w, x1, 128, fp2in, tot, 384);
  }
  gemmH(fp2in, fp2l1, fp2h, B * N1, 1);
  gemmH(fp2h, fp2l2, fp2o, B * N1, 0);

  // ---- FP1 (k=3, pos1 -> pos) ----
  knn3_kernel<<<(B * NPTS + 255) / 256, 256, 0, stream>>>(pos1, pos, k1i, k1w, B * NPTS,
                                                          NPTS, N1);
  {
    const long long tot = (long long)B * NPTS * 160;
    interp_concat_kernel<<<(int)((tot + 255) / 256), 256, 0, stream>>>(
        fp2o, 128, k1i, k1w, x0h, 3, fp1in, tot, 160);
  }
  gemmH(fp1in, fp1l1, fp1a, B * NPTS, 1);
  gemmH(fp1a, fp1l2, fp1b, B * NPTS, 1);
  gemmH(fp1b, fp1l3, fp1c, B * NPTS, 0);

  // ---- head + log_softmax ----
  gemmH(fp1c, hd1, hh1, B * NPTS, 1);
  gemmH(hh1, hd2, hh2, B * NPTS, 1);
  gemmF(hh2, hd3, logit, B * NPTS, 0);
  logsoftmax_kernel<<<(B * NPTS + 255) / 256, 256, 0, stream>>>(logit, (float*)d_out,
                                                                B * NPTS);
}
